// MsaHmmLayer_60206851555803
// MI455X (gfx1250) — matrix operand order repaired and checked by reading, not compiled, not run on hardware
//
#include <hip/hip_runtime.h>

typedef __attribute__((ext_vector_type(16))) _Float16 v16h;
typedef __attribute__((ext_vector_type(8)))  _Float16 v8h;
typedef __attribute__((ext_vector_type(8)))  float    v8f;

#define HMM_B 64
#define HMM_L 2048
#define HMM_Q 256
#define HMM_S 26
#define HMM_EPS 1e-16f
#define POST_ELEMS ((size_t)HMM_B * (size_t)HMM_L * (size_t)HMM_Q)

#define LN2_F    0.6931471805599453f
#define LOG2E_F  1.4426950408889634f

// Dynamic-LDS layout (bytes). 320KB/WGP on CDNA5 lets us keep exp(A) f16 resident.
#define OFF_AMAT   0          // 256*256 f16 = 131072  (exp(A), layout per kernel)
#define OFF_EB     131072     // 256*32  f16 = 16384   (exp(B), S padded to 32)
#define OFF_STATE  147456     // 16*256  f32 = 16384   (alpha or beta tile)
#define OFF_PQ     163840     // 16*256  f16 = 8192    (rescaled p or q activations)
#define OFF_INS    172032     // 16*32   f16 = 1024    (input tile for one timestep)
#define OFF_RED    173056     // 16*16   f32 = 1024    (max/sum reduce partials)
#define OFF_MROW   174080     // 16      f32 = 64      (per-row max)
#define OFF_LLK    174144     // 16      f32 = 64      (loglik, bwd only)
#define SMEM_FWD   174144
#define SMEM_BWD   174208

// raw hardware transcendentals (no denorm guards -- operands are rescaled to sane range)
__device__ __forceinline__ float fast_log(float x) {        // ln(x)
  return __builtin_amdgcn_logf(x) * LN2_F;                  // v_log_f32 (log2) + mul
}
__device__ __forceinline__ float fast_log_add(float x, float add) {  // ln(x) + add
  return fmaf(__builtin_amdgcn_logf(x), LN2_F, add);        // v_log_f32 + fma
}
__device__ __forceinline__ float fast_exp_sub(float x, float m2) {   // exp(x - m), m2 = m*log2e
  return __builtin_amdgcn_exp2f(fmaf(x, LOG2E_F, -m2));     // fma + v_exp_f32
}

__device__ __forceinline__ v8f wmma_f16(v16h a, v16h b, v8f c) {
  return __builtin_amdgcn_wmma_f32_16x16x32_f16(false, a, false, b, (short)0, c,
                                                false, false);
}

// A-operand fragment (16xK tile, 16-bit): lanes 0-15 row M=lane hold K {k0..k0+7, k0+16..k0+23};
// lanes 16-31 row M=lane-16 hold K {k0+8..k0+15, k0+24..k0+31}.
__device__ __forceinline__ v16h load_fragA(const _Float16* buf, int stride, int k0) {
  int lane = threadIdx.x & 31;
  const _Float16* p = buf + (lane & 15) * stride + k0 + ((lane >> 4) << 3);
  v8h lo = *(const v8h*)p;
  v8h hh = *(const v8h*)(p + 16);
  v16h a;
#pragma unroll
  for (int i = 0; i < 8; ++i) { a[i] = lo[i]; a[i + 8] = hh[i]; }
  return a;
}

// B-operand fragment (Kx16 tile, 16-bit): lane&15 = column n; lanes 0-15 hold K {k0..k0+7, k0+16..k0+23},
// lanes 16-31 hold K {k0+8..k0+15, k0+24..k0+31} (matches A). buf stores buf[n][k], contraction contiguous.
__device__ __forceinline__ v16h load_fragB(const _Float16* buf, int n0, int stride, int k0) {
  int lane = threadIdx.x & 31;
  const _Float16* p = buf + (n0 + (lane & 15)) * stride + k0 + ((lane >> 4) << 3);
  v8h lo = *(const v8h*)p;
  v8h hh = *(const v8h*)(p + 16);
  v16h b;
#pragma unroll
  for (int i = 0; i < 8; ++i) { b[i] = lo[i]; b[i + 8] = hh[i]; }
  return b;
}

// ---------------- forward: alpha recursion + loglik -------------------------
__global__ __launch_bounds__(256) void hmm_fwd(const float* __restrict__ inputs,
                                               const float* __restrict__ logA,
                                               const float* __restrict__ logpi,
                                               const float* __restrict__ logB,
                                               float* __restrict__ out) {
  extern __shared__ char smem[];
  _Float16* AT   = (_Float16*)(smem + OFF_AMAT);   // AT[i][j] = exp(logA[j][i])
  _Float16* eB   = (_Float16*)(smem + OFF_EB);     // eB[q][s] = exp(logB[q][s])
  float*    alph = (float*)   (smem + OFF_STATE);
  _Float16* pbuf = (_Float16*)(smem + OFF_PQ);
  _Float16* ins  = (_Float16*)(smem + OFF_INS);
  float*    red  = (float*)   (smem + OFF_RED);
  float*    mrow = (float*)   (smem + OFF_MROW);

  const int tid  = threadIdx.x;
  const int b0   = blockIdx.x * 16;
  const int wave = tid >> 5;
  const int lane = tid & 31;
  const int hi   = lane >> 4;
  const int ln   = lane & 15;
  const int n0a  = wave * 32;
  const int n0b  = n0a + 16;
  const int srow = tid >> 4;   // staging / streaming row
  const int scol = tid & 15;
  const int rrow = tid & 15;   // reduce row
  const int rseg = tid >> 4;

  for (int idx = tid; idx < HMM_Q * HMM_Q; idx += 256) {
    int i = idx >> 8, j = idx & 255;
    AT[i * HMM_Q + j] = (_Float16)__builtin_amdgcn_exp2f(logA[j * HMM_Q + i] * LOG2E_F);
  }
  for (int idx = tid; idx < HMM_Q * 32; idx += 256) {
    int q = idx >> 5, s = idx & 31;
    eB[idx] = (s < HMM_S)
                  ? (_Float16)__builtin_amdgcn_exp2f(logB[q * HMM_S + s] * LOG2E_F)
                  : (_Float16)0.0f;
  }

  auto stage_inputs = [&](int t) {
    const float* ip = inputs + ((size_t)(b0 + srow) * HMM_L + t) * HMM_S;
    ins[srow * 32 + scol] = (scol < HMM_S) ? (_Float16)ip[scol] : (_Float16)0.0f;
    int c2 = scol + 16;
    ins[srow * 32 + c2] = (c2 < HMM_S) ? (_Float16)ip[c2] : (_Float16)0.0f;
  };

  // max-reduce alpha_t -> mrow, build p = exp(alpha-m) f16, stream alpha_t to d_out
  auto reduce_and_emit = [&](int t) {
    {
      const float4* ar = (const float4*)(alph + rrow * HMM_Q + rseg * 16);
      float4 a0 = ar[0], a1 = ar[1], a2 = ar[2], a3 = ar[3];
      float mx = fmaxf(fmaxf(fmaxf(a0.x, a0.y), fmaxf(a0.z, a0.w)),
                       fmaxf(fmaxf(a1.x, a1.y), fmaxf(a1.z, a1.w)));
      mx = fmaxf(mx, fmaxf(fmaxf(fmaxf(a2.x, a2.y), fmaxf(a2.z, a2.w)),
                           fmaxf(fmaxf(a3.x, a3.y), fmaxf(a3.z, a3.w))));
      red[rrow * 16 + rseg] = mx;
    }
    __syncthreads();
    if (tid < 16) {
      const float4* rr = (const float4*)(red + tid * 16);
      float4 a0 = rr[0], a1 = rr[1], a2 = rr[2], a3 = rr[3];
      float mx = fmaxf(fmaxf(fmaxf(a0.x, a0.y), fmaxf(a0.z, a0.w)),
                       fmaxf(fmaxf(a1.x, a1.y), fmaxf(a1.z, a1.w)));
      mx = fmaxf(mx, fmaxf(fmaxf(fmaxf(a2.x, a2.y), fmaxf(a2.z, a2.w)),
                           fmaxf(fmaxf(a3.x, a3.y), fmaxf(a3.z, a3.w))));
      mrow[tid] = mx;
    }
    __syncthreads();
    {
      float m2 = mrow[srow] * LOG2E_F;
      const float4* av = (const float4*)(alph + srow * HMM_Q + scol * 16);
      float4 a0 = av[0], a1 = av[1], a2 = av[2], a3 = av[3];
      v8h h0, h1;
      h0[0] = (_Float16)fast_exp_sub(a0.x, m2); h0[1] = (_Float16)fast_exp_sub(a0.y, m2);
      h0[2] = (_Float16)fast_exp_sub(a0.z, m2); h0[3] = (_Float16)fast_exp_sub(a0.w, m2);
      h0[4] = (_Float16)fast_exp_sub(a1.x, m2); h0[5] = (_Float16)fast_exp_sub(a1.y, m2);
      h0[6] = (_Float16)fast_exp_sub(a1.z, m2); h0[7] = (_Float16)fast_exp_sub(a1.w, m2);
      h1[0] = (_Float16)fast_exp_sub(a2.x, m2); h1[1] = (_Float16)fast_exp_sub(a2.y, m2);
      h1[2] = (_Float16)fast_exp_sub(a2.z, m2); h1[3] = (_Float16)fast_exp_sub(a2.w, m2);
      h1[4] = (_Float16)fast_exp_sub(a3.x, m2); h1[5] = (_Float16)fast_exp_sub(a3.y, m2);
      h1[6] = (_Float16)fast_exp_sub(a3.z, m2); h1[7] = (_Float16)fast_exp_sub(a3.w, m2);
      *(v8h*)(pbuf + srow * HMM_Q + scol * 16)     = h0;
      *(v8h*)(pbuf + srow * HMM_Q + scol * 16 + 8) = h1;
      float4* g = (float4*)(out + ((size_t)(b0 + srow) * HMM_L + t) * HMM_Q + scol * 16);
      g[0] = a0; g[1] = a1; g[2] = a2; g[3] = a3;
    }
  };

  // ---- t = 0: alpha0 = log_pi + log(emit0 + eps) ----
  stage_inputs(0);
  __syncthreads();
  {
    v16h af = load_fragA(ins, 32, 0);
    v8f ye0 = {}, ye1 = {};
    ye0 = wmma_f16(af, load_fragB(eB, n0a, 32, 0), ye0);
    ye1 = wmma_f16(af, load_fragB(eB, n0b, 32, 0), ye1);
    float lp0 = logpi[n0a + ln];
    float lp1 = logpi[n0b + ln];
#pragma unroll
    for (int r = 0; r < 8; ++r) {
      int row = hi * 8 + r;
      alph[row * HMM_Q + n0a + ln] = fast_log_add(ye0[r] + HMM_EPS, lp0);
      alph[row * HMM_Q + n0b + ln] = fast_log_add(ye1[r] + HMM_EPS, lp1);
    }
  }
  __syncthreads();
  reduce_and_emit(0);

  // ---- t = 1 .. L-1 ----
  for (int t = 1; t < HMM_L; ++t) {
    stage_inputs(t);
    __syncthreads();  // ins/pbuf/mrow ready; prior alpha readers done
    v16h af = load_fragA(ins, 32, 0);
    v8f ye0 = {}, ye1 = {};
    ye0 = wmma_f16(af, load_fragB(eB, n0a, 32, 0), ye0);
    ye1 = wmma_f16(af, load_fragB(eB, n0b, 32, 0), ye1);
    v8f yt0 = {}, yt1 = {};
#pragma unroll
    for (int kc = 0; kc < 8; ++kc) {
      v16h ap = load_fragA(pbuf, HMM_Q, kc * 32);
      yt0 = wmma_f16(ap, load_fragB(AT, n0a, HMM_Q, kc * 32), yt0);
      yt1 = wmma_f16(ap, load_fragB(AT, n0b, HMM_Q, kc * 32), yt1);
    }
    float mr[8];
#pragma unroll
    for (int r = 0; r < 8; ++r) mr[r] = mrow[hi * 8 + r];
    // alpha_t = m + log(y_trans * (emit+eps))  -- single v_log_f32 per element
#pragma unroll
    for (int r = 0; r < 8; ++r) {
      int row = hi * 8 + r;
      alph[row * HMM_Q + n0a + ln] = fast_log_add(yt0[r] * (ye0[r] + HMM_EPS), mr[r]);
      alph[row * HMM_Q + n0b + ln] = fast_log_add(yt1[r] * (ye1[r] + HMM_EPS), mr[r]);
    }
    __syncthreads();  // alpha_t complete
    reduce_and_emit(t);
  }

  // ---- loglik = m + log(sum exp(alpha_{L-1} - m)) ----
  __syncthreads();
  {
    float m2 = mrow[rrow] * LOG2E_F;
    const float4* ar = (const float4*)(alph + rrow * HMM_Q + rseg * 16);
    float s = 0.0f;
#pragma unroll
    for (int v = 0; v < 4; ++v) {
      float4 a = ar[v];
      s += fast_exp_sub(a.x, m2) + fast_exp_sub(a.y, m2) +
           fast_exp_sub(a.z, m2) + fast_exp_sub(a.w, m2);
    }
    red[rrow * 16 + rseg] = s;
  }
  __syncthreads();
  if (tid < 16) {
    const float4* rr = (const float4*)(red + tid * 16);
    float s = 0.0f;
#pragma unroll
    for (int v = 0; v < 4; ++v) {
      float4 a = rr[v];
      s += a.x + a.y + a.z + a.w;
    }
    out[POST_ELEMS + b0 + tid] = fast_log_add(s, mrow[tid]);
  }
}

// -------- backward: beta recursion fused with post = alpha + beta - loglik --------
__global__ __launch_bounds__(256) void hmm_bwd(const float* __restrict__ inputs,
                                               const float* __restrict__ logA,
                                               const float* __restrict__ logB,
                                               float* __restrict__ out) {
  extern __shared__ char smem[];
  _Float16* AR   = (_Float16*)(smem + OFF_AMAT);   // AR[i][j] = exp(logA[i][j])
  _Float16* eB   = (_Float16*)(smem + OFF_EB);
  float*    beta = (float*)   (smem + OFF_STATE);
  _Float16* qbuf = (_Float16*)(smem + OFF_PQ);
  _Float16* ins  = (_Float16*)(smem + OFF_INS);
  float*    red  = (float*)   (smem + OFF_RED);
  float*    mrow = (float*)   (smem + OFF_MROW);
  float*    llk  = (float*)   (smem + OFF_LLK);

  const int tid  = threadIdx.x;
  const int b0   = blockIdx.x * 16;
  const int wave = tid >> 5;
  const int lane = tid & 31;
  const int hi   = lane >> 4;
  const int ln   = lane & 15;
  const int n0a  = wave * 32;
  const int n0b  = n0a + 16;
  const int srow = tid >> 4;
  const int scol = tid & 15;
  const int rrow = tid & 15;
  const int rseg = tid >> 4;

  for (int idx = tid; idx < HMM_Q * HMM_Q; idx += 256)
    AR[idx] = (_Float16)__builtin_amdgcn_exp2f(logA[idx] * LOG2E_F);
  for (int idx = tid; idx < HMM_Q * 32; idx += 256) {
    int q = idx >> 5, s = idx & 31;
    eB[idx] = (s < HMM_S)
                  ? (_Float16)__builtin_amdgcn_exp2f(logB[q * HMM_S + s] * LOG2E_F)
                  : (_Float16)0.0f;
  }
  for (int idx = tid; idx < 16 * HMM_Q; idx += 256) beta[idx] = 0.0f;
  if (tid < 16) { mrow[tid] = 0.0f; llk[tid] = out[POST_ELEMS + b0 + tid]; }
  __syncthreads();

  // post at t = L-1: beta = 0 -> post = alpha - loglik (in place)
  {
    float ll = llk[srow];
    float4* g = (float4*)(out + ((size_t)(b0 + srow) * HMM_L + (HMM_L - 1)) * HMM_Q +
                          scol * 16);
#pragma unroll
    for (int v = 0; v < 4; ++v) {
      float4 x = g[v];
      x.x -= ll; x.y -= ll; x.z -= ll; x.w -= ll;
      g[v] = x;
    }
  }

  for (int t = HMM_L - 2; t >= 0; --t) {
    // stage inputs at t+1 (backward uses emission at t+1; no log needed:
    // exp(log_e + beta - m) = (emit+eps)*exp(beta-m))
    {
      const float* ip = inputs + ((size_t)(b0 + srow) * HMM_L + (t + 1)) * HMM_S;
      ins[srow * 32 + scol] = (scol < HMM_S) ? (_Float16)ip[scol] : (_Float16)0.0f;
      int c2 = scol + 16;
      ins[srow * 32 + c2] = (c2 < HMM_S) ? (_Float16)ip[c2] : (_Float16)0.0f;
    }
    __syncthreads();  // B1
    v16h af = load_fragA(ins, 32, 0);
    v8f ye0 = {}, ye1 = {};
    ye0 = wmma_f16(af, load_fragB(eB, n0a, 32, 0), ye0);
    ye1 = wmma_f16(af, load_fragB(eB, n0b, 32, 0), ye1);
    float mr[8];
#pragma unroll
    for (int r = 0; r < 8; ++r) mr[r] = mrow[hi * 8 + r];
#pragma unroll
    for (int r = 0; r < 8; ++r) {
      int row = hi * 8 + r;
      float m2 = mr[r] * LOG2E_F;
      float q0 = (ye0[r] + HMM_EPS) * fast_exp_sub(beta[row * HMM_Q + n0a + ln], m2);
      float q1 = (ye1[r] + HMM_EPS) * fast_exp_sub(beta[row * HMM_Q + n0b + ln], m2);
      qbuf[row * HMM_Q + n0a + ln] = (_Float16)q0;
      qbuf[row * HMM_Q + n0b + ln] = (_Float16)q1;
    }
    __syncthreads();  // B2: qbuf ready, beta reads done
    v8f yt0 = {}, yt1 = {};
#pragma unroll
    for (int kc = 0; kc < 8; ++kc) {
      v16h aq = load_fragA(qbuf, HMM_Q, kc * 32);
      yt0 = wmma_f16(aq, load_fragB(AR, n0a, HMM_Q, kc * 32), yt0);
      yt1 = wmma_f16(aq, load_fragB(AR, n0b, HMM_Q, kc * 32), yt1);
    }
    // beta_t = m' + log(sum_j exp(A)[i,j] * q[j])
#pragma unroll
    for (int r = 0; r < 8; ++r) {
      int row = hi * 8 + r;
      beta[row * HMM_Q + n0a + ln] = fast_log_add(yt0[r] + 1e-30f, mr[r]);
      beta[row * HMM_Q + n0b + ln] = fast_log_add(yt1[r] + 1e-30f, mr[r]);
    }
    __syncthreads();  // B3: beta_t complete
    {
      const float4* br = (const float4*)(beta + rrow * HMM_Q + rseg * 16);
      float4 a0 = br[0], a1 = br[1], a2 = br[2], a3 = br[3];
      float mx = fmaxf(fmaxf(fmaxf(a0.x, a0.y), fmaxf(a0.z, a0.w)),
                       fmaxf(fmaxf(a1.x, a1.y), fmaxf(a1.z, a1.w)));
      mx = fmaxf(mx, fmaxf(fmaxf(fmaxf(a2.x, a2.y), fmaxf(a2.z, a2.w)),
                           fmaxf(fmaxf(a3.x, a3.y), fmaxf(a3.z, a3.w))));
      red[rrow * 16 + rseg] = mx;
    }
    __syncthreads();  // B4
    if (tid < 16) {
      const float4* rr = (const float4*)(red + tid * 16);
      float4 a0 = rr[0], a1 = rr[1], a2 = rr[2], a3 = rr[3];
      float mx = fmaxf(fmaxf(fmaxf(a0.x, a0.y), fmaxf(a0.z, a0.w)),
                       fmaxf(fmaxf(a1.x, a1.y), fmaxf(a1.z, a1.w)));
      mx = fmaxf(mx, fmaxf(fmaxf(fmaxf(a2.x, a2.y), fmaxf(a2.z, a2.w)),
                           fmaxf(fmaxf(a3.x, a3.y), fmaxf(a3.z, a3.w))));
      mrow[tid] = mx;
    }
    __syncthreads();  // B5
    // post[t] = alpha[t] + beta_t - loglik (in-place float4 RMW on d_out)
    {
      float ll = llk[srow];
      const float4* br = (const float4*)(beta + srow * HMM_Q + scol * 16);
      float4* g = (float4*)(out + ((size_t)(b0 + srow) * HMM_L + t) * HMM_Q + scol * 16);
#pragma unroll
      for (int v = 0; v < 4; ++v) {
        float4 b = br[v];
        float4 x = g[v];
        x.x += b.x - ll; x.y += b.y - ll;
        x.z += b.z - ll; x.w += b.w - ll;
        g[v] = x;
      }
    }
  }
}

extern "C" void kernel_launch(void* const* d_in, const int* in_sizes, int n_in,
                              void* d_out, int out_size, void* d_ws, size_t ws_size,
                              hipStream_t stream) {
  (void)in_sizes; (void)n_in; (void)out_size; (void)d_ws; (void)ws_size;
  const float* inputs = (const float*)d_in[0];
  const float* logA   = (const float*)d_in[1];
  const float* logpi  = (const float*)d_in[2];
  const float* logB   = (const float*)d_in[3];
  float* out = (float*)d_out;

  hmm_fwd<<<dim3(HMM_B / 16), dim3(256), SMEM_FWD, stream>>>(inputs, logA, logpi,
                                                             logB, out);
  hmm_bwd<<<dim3(HMM_B / 16), dim3(256), SMEM_BWD, stream>>>(inputs, logA, logB, out);
}
